// GTM_SM_21088289423529
// MI455X (gfx1250) — matrix operand order repaired and checked by reading, not compiled, not run on hardware
//
#include <hip/hip_runtime.h>
#include <hip/hip_bf16.h>
#include <math.h>

#define B_N   256
#define T_N   288
#define OBS_N 256
#define TP_N  32
#define EPSV  1e-5f

typedef __attribute__((ext_vector_type(16))) _Float16 v16h;
typedef __attribute__((ext_vector_type(8)))  float    v8f;

__device__ __forceinline__ float sigmoidf_(float x) { return 1.0f / (1.0f + expf(-x)); }

// ---- WMMA fragment loaders (wave32, contiguous k order used the same in A and B) ----
// A: 16x32 f16 row-major [M][K]. lanes0-15: row=lane, K {k0..k0+15};
// lanes16-31: row=lane-16, K {k0+16..k0+31}.  -> 2x ds_load_b128 per lane.
__device__ __forceinline__ v16h ldA16(const _Float16* A, int lda, int m0, int k0, int lane) {
  const int r  = m0 + (lane & 15);
  const int kb = (lane < 16) ? 0 : 16;
  const _Float16* p = A + r * lda + k0 + kb;
  v16h a;
#pragma unroll
  for (int j = 0; j < 8; ++j) a[j] = p[j];
#pragma unroll
  for (int j = 0; j < 8; ++j) a[8 + j] = p[8 + j];
  return a;
}
// B staged pre-swizzled in fragment order: Bsw[tile][lane][j], 32B contiguous per lane
// (fragment element j = B[k0 + ((lane<16)?0:16) + j][n0 + (lane&15)]).  -> 2x ds_load_b128.
__device__ __forceinline__ v16h ldBsw(const _Float16* Bsw, int tile, int lane) {
  const _Float16* p = Bsw + (tile * 32 + lane) * 16;
  v16h b;
#pragma unroll
  for (int j = 0; j < 16; ++j) b[j] = p[j];
  return b;
}

__global__ void k_zero_nll(float* p) { if (threadIdx.x == 0) *p = 0.0f; }

// ---------------- Kernel 1: sequential state recurrence (latency-bound path) ----------------
__global__ void k_recur(const float* __restrict__ act, const float* __restrict__ s0in,
                        const float* __restrict__ nobs, const float* __restrict__ nprd,
                        const float* __restrict__ W_a, const float* __restrict__ Ws1,
                        const float* __restrict__ bs1, const float* __restrict__ Ws2,
                        const float* __restrict__ bs2,
                        float* __restrict__ st_obs, float* __restrict__ st_pred) {
  const int b = threadIdx.x;  // one sample per lane, 256 threads = 8 waves
  float wa0[5], wa1[5], w1a[5], w1b[5], b1[5], w20[5], w21[5];
#pragma unroll
  for (int a = 0; a < 5; ++a) {
    wa0[a] = W_a[a];        wa1[a] = W_a[5 + a];
    w1a[a] = Ws1[a * 2];    w1b[a] = Ws1[a * 2 + 1];  b1[a] = bs1[a];
    w20[a] = Ws2[a];        w21[a] = Ws2[5 + a];
  }
  const float bb0 = bs2[0], bb1 = bs2[1];
  float sv0 = s0in[b * 2 + 0], sv1 = s0in[b * 2 + 1];
  st_obs[b * 2 + 0] = sv0;  st_obs[b * 2 + 1] = sv1;
  for (int t = 1; t < T_N; ++t) {
    float u0 = 0.f, u1 = 0.f;
#pragma unroll
    for (int a = 0; a < 5; ++a) {
      float av = act[(b * 5 + a) * T_N + t];
      u0 += wa0[a] * av;  u1 += wa1[a] * av;
    }
    float v0 = sv0 + u0, v1 = sv1 + u1;
    float m0 = bb0, m1 = bb1;
#pragma unroll
    for (int j = 0; j < 5; ++j) {
      float h = tanhf(w1a[j] * v0 + w1b[j] * v1 + b1[j]);
      m0 += w20[j] * h;  m1 += w21[j] * h;
    }
    m0 = sigmoidf_(m0);  m1 = sigmoidf_(m1);
    float n0, n1;
    if (t < OBS_N) { n0 = nobs[((t - 1) * B_N + b) * 2]; n1 = nobs[((t - 1) * B_N + b) * 2 + 1]; }
    else           { n0 = nprd[((t - OBS_N) * B_N + b) * 2]; n1 = nprd[((t - OBS_N) * B_N + b) * 2 + 1]; }
    sv0 += u0 * m0 + n0;
    sv1 += u1 * m1 + n1;
    if (t < OBS_N) { st_obs[(t * B_N + b) * 2] = sv0; st_obs[(t * B_N + b) * 2 + 1] = sv1; }
    else { st_pred[((t - OBS_N) * B_N + b) * 2] = sv0; st_pred[((t - OBS_N) * B_N + b) * 2 + 1] = sv1; }
  }
}

// ---------------- Kernel 2: glimpse encoder, one workgroup per timestep ----------------
// conv1+pool streamed per non-overlapping 4x4 block (no patch LDS) -> BN1 folded into conv2
// weights -> WMMA GEMM (256x64 @ 64x32) -> BN2 folded into z-head weights ->
// WMMA GEMM (256x32 @ 32x32) -> zm / exp(zs).  LDS: 57KB -> 5 WGs/WGP.
__global__ void k_encoder(const float* __restrict__ x, const int* __restrict__ pos,
                          const float* __restrict__ Wc1, const float* __restrict__ bc1,
                          const float* __restrict__ g1, const float* __restrict__ be1,
                          const float* __restrict__ Wc2, const float* __restrict__ bc2,
                          const float* __restrict__ g2, const float* __restrict__ be2,
                          const float* __restrict__ Wzm, const float* __restrict__ bzm,
                          const float* __restrict__ Wzs, const float* __restrict__ bzs,
                          float* __restrict__ zm_obs, float* __restrict__ zs_obs,
                          float* __restrict__ zm_pred, float* __restrict__ zs_pred) {
  extern __shared__ char smem[];
  _Float16* A1   = (_Float16*)smem;                  // [256][64]  32KB pooled conv1
  _Float16* h2s  = (_Float16*)(smem + 32768);        // [256][32]  16KB relu(conv2')
  _Float16* Bw   = (_Float16*)(smem + 49152);        // swizzled [4 tiles][32][16] 4KB
  _Float16* Bz   = (_Float16*)(smem + 53248);        // swizzled [2 tiles][32][16] 2KB
  float* sWc1    = (float*)(smem + 55296);           // 192 floats conv1 weights
  float* stats   = (float*)(smem + 56064);           // 256 floats
  float* s1m = stats;        float* s1v = stats + 16;
  float* a1f = stats + 32;   float* d1f = stats + 48;
  float* s2m = stats + 64;   float* s2v = stats + 96;
  float* a2f = stats + 128;  float* d2f = stats + 160;
  float* bias2s = stats + 192;  float* biaszs = stats + 224;

  const int t = blockIdx.x;
  const int tid = threadIdx.x;
  const int lane = tid & 31;
  const int wv = tid >> 5;

  if (tid < 128) stats[tid] = 0.0f;
  for (int i = tid; i < 192; i += 256) sWc1[i] = Wc1[i];
  __syncthreads();

  // 1) conv1 (16oc,2x2,s2)+relu+2x2 maxpool, streamed per non-overlapping 4x4 block
  {
    const int b = tid;
    const int r0 = 3 * pos[(b * 2 + 0) * T_N + t];
    const int c0 = 3 * pos[(b * 2 + 1) * T_N + t];
    const float* xb = x + (size_t)b * 3 * 32 * 32;
    __builtin_prefetch(xb + (r0 * 32 + c0), 0, 0);  // global_prefetch_b8: gathered glimpse rows
    float lsum[16], lsq[16];
#pragma unroll
    for (int o = 0; o < 16; ++o) { lsum[o] = 0.f; lsq[o] = 0.f; }
#pragma unroll
    for (int py = 0; py < 2; ++py)
      for (int px = 0; px < 2; ++px) {
        float blk[48];  // 3ch x 4x4 input block feeding this pooled cell
#pragma unroll
        for (int ch = 0; ch < 3; ++ch)
          for (int iy = 0; iy < 4; ++iy)
            for (int jx = 0; jx < 4; ++jx)
              blk[ch * 16 + iy * 4 + jx] = xb[(ch * 32 + r0 + py * 4 + iy) * 32 + (c0 + px * 4 + jx)];
        for (int oc = 0; oc < 16; ++oc) {
          float mx = -1e30f;
#pragma unroll
          for (int sy = 0; sy < 2; ++sy)
            for (int sx = 0; sx < 2; ++sx) {
              float acc = bc1[oc];
#pragma unroll
              for (int ch = 0; ch < 3; ++ch)
                for (int ky = 0; ky < 2; ++ky)
                  for (int kx = 0; kx < 2; ++kx)
                    acc += sWc1[oc * 12 + ch * 4 + ky * 2 + kx] *
                           blk[ch * 16 + (sy * 2 + ky) * 4 + (sx * 2 + kx)];
              acc = fmaxf(acc, 0.f);
              mx = fmaxf(mx, acc);
            }
          A1[b * 64 + oc * 4 + py * 2 + px] = (_Float16)mx;
          lsum[oc] += mx;  lsq[oc] += mx * mx;
        }
      }
    for (int oc = 0; oc < 16; ++oc) {
      atomicAdd(&s1m[oc], lsum[oc]);
      atomicAdd(&s1v[oc], lsq[oc]);
    }
  }
  __syncthreads();
  if (tid < 16) {  // BN1 scale/shift (N = 256*4)
    float m = s1m[tid] * (1.0f / 1024.0f);
    float v = s1v[tid] * (1.0f / 1024.0f) - m * m;
    float a = g1[tid] * rsqrtf(v + EPSV);
    a1f[tid] = a;  d1f[tid] = be1[tid] - m * a;
  }
  __syncthreads();
  // 2) fold BN1 into conv2 weights, stage swizzled (tile = kt*2+nt)
  if (tid < 64) {
    const int k = tid, ch = k >> 2, ky = (k >> 1) & 1, kx = k & 1;
    const float a = a1f[ch];
    const int kt = k >> 5, jj = k & 15, half = k & 16;
    for (int c = 0; c < 32; ++c) {
      const int tile = kt * 2 + (c >> 4);
      const int lanei = half + (c & 15);
      Bw[(tile * 32 + lanei) * 16 + jj] = (_Float16)(Wc2[((c * 16 + ch) * 2 + ky) * 2 + kx] * a);
    }
  } else if (tid < 96) {
    const int c = tid - 64;
    float s = bc2[c];
    for (int k = 0; k < 64; ++k) {
      const int ch = k >> 2, ky = (k >> 1) & 1, kx = k & 1;
      s += Wc2[((c * 16 + ch) * 2 + ky) * 2 + kx] * d1f[ch];
    }
    bias2s[c] = s;
  }
  __syncthreads();
  // 3) WMMA GEMM1: [256x64]f16 @ [64x32]f16 -> relu -> h2s
  for (int mi = 0; mi < 2; ++mi) {
    const int mt = wv * 2 + mi;
    for (int nt = 0; nt < 2; ++nt) {
      v8f acc = {0.f, 0.f, 0.f, 0.f, 0.f, 0.f, 0.f, 0.f};
#pragma unroll
      for (int kt = 0; kt < 2; ++kt) {
        v16h a = ldA16(A1, 64, mt * 16, kt * 32, lane);
        v16h b = ldBsw(Bw, kt * 2 + nt, lane);
        acc = __builtin_amdgcn_wmma_f32_16x16x32_f16(false, a, false, b, (short)0, acc, false, false);
      }
      const int n = nt * 16 + (lane & 15);
      const int mb = mt * 16 + ((lane < 16) ? 0 : 8);
      const float bz = bias2s[n];
#pragma unroll
      for (int r = 0; r < 8; ++r) {
        float v = fmaxf(acc[r] + bz, 0.f);
        h2s[(mb + r) * 32 + n] = (_Float16)v;
      }
    }
  }
  __syncthreads();
  // 4) BN2 stats over batch
  {
    const int c = tid & 31, part = tid >> 5;
    float ls = 0.f, lq = 0.f;
    for (int b = part * 32; b < part * 32 + 32; ++b) {
      float v = (float)h2s[b * 32 + c];
      ls += v;  lq += v * v;
    }
    atomicAdd(&s2m[c], ls);
    atomicAdd(&s2v[c], lq);
  }
  __syncthreads();
  if (tid < 32) {
    float m = s2m[tid] * (1.0f / 256.0f);
    float v = s2v[tid] * (1.0f / 256.0f) - m * m;
    float a = g2[tid] * rsqrtf(v + EPSV);
    a2f[tid] = a;  d2f[tid] = be2[tid] - m * a;
  }
  __syncthreads();
  // 5) fold BN2 into z-head weights, swizzled (cols 0..15 mean head, 16..31 log-std head)
  if (tid < 32) {
    const int k = tid;  const float a = a2f[k];
    const int jj = k & 15, half = k & 16;
    for (int n = 0; n < 32; ++n) {
      float w = (n < 16) ? Wzm[n * 32 + k] : Wzs[(n - 16) * 32 + k];
      const int tile = n >> 4;
      const int lanei = half + (n & 15);
      Bz[(tile * 32 + lanei) * 16 + jj] = (_Float16)(w * a);
    }
  } else if (tid < 64) {
    const int n = tid - 32;
    float s = (n < 16) ? bzm[n] : bzs[n - 16];
    for (int k = 0; k < 32; ++k) {
      float w = (n < 16) ? Wzm[n * 32 + k] : Wzs[(n - 16) * 32 + k];
      s += w * d2f[k];
    }
    biaszs[n] = s;
  }
  __syncthreads();
  // 6) WMMA GEMM2: [256x32] @ [32x32] -> zm / exp(zs) straight to HBM
  float* zm = (t < OBS_N) ? (zm_obs + (size_t)t * B_N * 16) : (zm_pred + (size_t)(t - OBS_N) * B_N * 16);
  float* zs = (t < OBS_N) ? (zs_obs + (size_t)t * B_N * 16) : (zs_pred + (size_t)(t - OBS_N) * B_N * 16);
  for (int mi = 0; mi < 2; ++mi) {
    const int mt = wv * 2 + mi;
    for (int nt = 0; nt < 2; ++nt) {
      v16h a = ldA16(h2s, 32, mt * 16, 0, lane);
      v16h b = ldBsw(Bz, nt, lane);
      v8f acc = {0.f, 0.f, 0.f, 0.f, 0.f, 0.f, 0.f, 0.f};
      acc = __builtin_amdgcn_wmma_f32_16x16x32_f16(false, a, false, b, (short)0, acc, false, false);
      const int n = nt * 16 + (lane & 15);
      const int mb = mt * 16 + ((lane < 16) ? 0 : 8);
      const float bz = biaszs[n];
#pragma unroll
      for (int r = 0; r < 8; ++r) {
        float v = acc[r] + bz;
        const int bi = mb + r;
        if (n < 16) zm[bi * 16 + n] = v;
        else        zs[bi * 16 + (n - 16)] = expf(v);
      }
    }
  }
}

// ---------------- Kernel 3: decoder + recon + NLL, one workgroup per pred timestep ----------------
__global__ void k_decoder(const float* __restrict__ zm_pred, const float* __restrict__ zs_pred,
                          const float* __restrict__ eps,
                          const float* __restrict__ Wd1, const float* __restrict__ bd1,
                          const float* __restrict__ gd1, const float* __restrict__ bed1,
                          const float* __restrict__ Wd2, const float* __restrict__ bd2,
                          const float* __restrict__ gd2, const float* __restrict__ bed2,
                          const float* __restrict__ Wt1, const float* __restrict__ bt1,
                          const float* __restrict__ gt1, const float* __restrict__ bet1,
                          const float* __restrict__ Wt2, const float* __restrict__ bt2,
                          const float* __restrict__ x, const int* __restrict__ pos,
                          float* __restrict__ xrec, float* __restrict__ nll) {
  extern __shared__ char smem[];
  _Float16* Az  = (_Float16*)smem;                 // [256][32] (K padded 16->32), dead after GEMM1
  _Float16* Bd1 = (_Float16*)(smem + 16384);       // swizzled [4 tiles][32][16]
  _Float16* h1d = (_Float16*)(smem + 20480);       // [256][64]
  _Float16* Bd2 = (_Float16*)(smem + 53248);       // swizzled [16 tiles][32][16]
  _Float16* t1b = (_Float16*)smem;                 // [256][256] convT1 out, reuses dead region
  _Float16* h2d = (_Float16*)(smem + 131072);      // [256][128]
  float* stats  = (float*)(smem + 196608);         // 640 floats
  float* sm = stats;         float* sv = stats + 128;
  float* af = stats + 256;   float* df = stats + 384;
  float* biasb = stats + 512;
  float* sWt1 = (float*)(smem + 199168);           // 4608 floats
  float* sWt2 = (float*)(smem + 217600);           // 192 floats

  const int t = blockIdx.x;
  const int tid = threadIdx.x;
  const int lane = tid & 31;
  const int wv = tid >> 5;

  stats[tid] = 0.0f;  // zero sm[128]+sv[128]
  for (int i = tid; i < 4608; i += 256) sWt1[i] = Wt1[i];
  for (int i = tid; i < 192;  i += 256) sWt2[i] = Wt2[i];

  // reparameterized z sample, K zero-padded to 32
  {
    const int b = tid;
    const float* zmr = zm_pred + ((size_t)t * B_N + b) * 16;
    const float* zsr = zs_pred + ((size_t)t * B_N + b) * 16;
    const float* er  = eps + ((size_t)t * B_N + b) * 16;
    for (int k = 0; k < 16; ++k) Az[b * 32 + k] = (_Float16)(zmr[k] + er[k] * zsr[k]);
    for (int k = 16; k < 32; ++k) Az[b * 32 + k] = (_Float16)0.0f;
  }
  if (tid < 32) {
    const int k = tid;
    const int jj = k & 15, half = k & 16;
    for (int n = 0; n < 64; ++n) {
      const int tile = n >> 4;
      const int lanei = half + (n & 15);
      Bd1[(tile * 32 + lanei) * 16 + jj] = (k < 16) ? (_Float16)Wd1[n * 16 + k] : (_Float16)0.0f;
    }
  }
  __syncthreads();
  // WMMA GEMM1: [256x32] @ [32x64] -> relu -> h1d
  for (int mi = 0; mi < 2; ++mi) {
    const int mt = wv * 2 + mi;
    for (int nt = 0; nt < 4; ++nt) {
      v16h a = ldA16(Az, 32, mt * 16, 0, lane);
      v16h b = ldBsw(Bd1, nt, lane);
      v8f acc = {0.f, 0.f, 0.f, 0.f, 0.f, 0.f, 0.f, 0.f};
      acc = __builtin_amdgcn_wmma_f32_16x16x32_f16(false, a, false, b, (short)0, acc, false, false);
      const int n = nt * 16 + (lane & 15);
      const int mb = mt * 16 + ((lane < 16) ? 0 : 8);
      const float bb = bd1[n];
#pragma unroll
      for (int r = 0; r < 8; ++r)
        h1d[(mb + r) * 64 + n] = (_Float16)fmaxf(acc[r] + bb, 0.f);
    }
  }
  __syncthreads();
  // BN(d1) stats over batch
  {
    const int f = tid & 63, part = tid >> 6;
    float ls = 0.f, lq = 0.f;
    for (int b = part * 64; b < part * 64 + 64; ++b) {
      float v = (float)h1d[b * 64 + f];
      ls += v;  lq += v * v;
    }
    atomicAdd(&sm[f], ls);
    atomicAdd(&sv[f], lq);
  }
  __syncthreads();
  if (tid < 64) {
    float m = sm[tid] * (1.f / 256.f), v = sv[tid] * (1.f / 256.f) - m * m;
    float a = gd1[tid] * rsqrtf(v + EPSV);
    af[tid] = a;  df[tid] = bed1[tid] - m * a;
    sm[tid] = 0.f;  sv[tid] = 0.f;  // re-zero for BN(d2)
  }
  __syncthreads();
  // fold BN(d1) into Wd2, stage swizzled (tile = kt*8+nt)
  if (tid < 64) {
    const int k = tid;  const float a = af[k];
    const int kt = k >> 5, jj = k & 15, half = k & 16;
    for (int n = 0; n < 128; ++n) {
      const int tile = kt * 8 + (n >> 4);
      const int lanei = half + (n & 15);
      Bd2[(tile * 32 + lanei) * 16 + jj] = (_Float16)(Wd2[n * 64 + k] * a);
    }
  } else if (tid < 192) {
    const int n = tid - 64;
    float s = bd2[n];
    for (int k = 0; k < 64; ++k) s += Wd2[n * 64 + k] * df[k];
    biasb[n] = s;
  }
  __syncthreads();
  // WMMA GEMM2: [256x64] @ [64x128] -> relu -> h2d
  for (int mi = 0; mi < 2; ++mi) {
    const int mt = wv * 2 + mi;
    for (int nt = 0; nt < 8; ++nt) {
      v8f acc = {0.f, 0.f, 0.f, 0.f, 0.f, 0.f, 0.f, 0.f};
#pragma unroll
      for (int kt = 0; kt < 2; ++kt) {
        v16h a = ldA16(h1d, 64, mt * 16, kt * 32, lane);
        v16h b = ldBsw(Bd2, kt * 8 + nt, lane);
        acc = __builtin_amdgcn_wmma_f32_16x16x32_f16(false, a, false, b, (short)0, acc, false, false);
      }
      const int n = nt * 16 + (lane & 15);
      const int mb = mt * 16 + ((lane < 16) ? 0 : 8);
      const float bb = biasb[n];
#pragma unroll
      for (int r = 0; r < 8; ++r)
        h2d[(mb + r) * 128 + n] = (_Float16)fmaxf(acc[r] + bb, 0.f);
    }
  }
  __syncthreads();
  // BN(d2) stats over batch (128 features)
  {
    const int f = tid & 127, part = tid >> 7;
    float ls = 0.f, lq = 0.f;
    for (int b = part * 128; b < part * 128 + 128; ++b) {
      float v = (float)h2d[b * 128 + f];
      ls += v;  lq += v * v;
    }
    atomicAdd(&sm[f], ls);
    atomicAdd(&sv[f], lq);
  }
  __syncthreads();
  if (tid < 128) {
    float m = sm[tid] * (1.f / 256.f), v = sv[tid] * (1.f / 256.f) - m * m;
    float a = gd2[tid] * rsqrtf(v + EPSV);
    af[tid] = a;  df[tid] = bed2[tid] - m * a;
    sm[tid] = 0.f;  sv[tid] = 0.f;  // re-zero for BN(t1)
  }
  __syncthreads();
  // apply BN(d2) in place, then convT1 (32->16, k3 s3 p1): exactly one (iy,ix) per output
  {
    const int b = tid;
    for (int n = 0; n < 128; ++n)
      h2d[b * 128 + n] = (_Float16)(af[n] * (float)h2d[b * 128 + n] + df[n]);
    for (int o = 0; o < 16; ++o) {
      const float bias = bt1[o];
      float ls = 0.f, lq = 0.f;
      for (int y = 0; y < 4; ++y) {
        const int iy = y >> 1, ky = y + 1 - 3 * iy;
        for (int xx = 0; xx < 4; ++xx) {
          const int ix = xx >> 1, kx = xx + 1 - 3 * ix;
          float acc = bias;
          for (int i = 0; i < 32; ++i)
            acc += (float)h2d[b * 128 + i * 4 + iy * 2 + ix] * sWt1[((i * 16 + o) * 3 + ky) * 3 + kx];
          acc = fmaxf(acc, 0.f);
          t1b[b * 256 + o * 16 + y * 4 + xx] = (_Float16)acc;
          ls += acc;  lq += acc * acc;
        }
      }
      atomicAdd(&sm[o], ls);
      atomicAdd(&sv[o], lq);
    }
  }
  __syncthreads();
  if (tid < 16) {  // BN(t1) over batch*spatial (N = 256*16)
    float m = sm[tid] * (1.f / 4096.f), v = sv[tid] * (1.f / 4096.f) - m * m;
    float a = gt1[tid] * rsqrtf(v + EPSV);
    af[tid] = a;  df[tid] = bet1[tid] - m * a;
  }
  __syncthreads();
  // apply BN(t1) in place, convT2 (16->3, k2 s2 p0) + sigmoid + recon store + NLL
  {
    const int b = tid;
    for (int idx = 0; idx < 256; ++idx) {
      const int o = idx >> 4;
      t1b[b * 256 + idx] = (_Float16)(af[o] * (float)t1b[b * 256 + idx] + df[o]);
    }
    const int r0 = 3 * pos[(b * 2 + 0) * T_N + (OBS_N + t)];
    const int c0 = 3 * pos[(b * 2 + 1) * T_N + (OBS_N + t)];
    const float* xb = x + (size_t)b * 3 * 32 * 32;
    __builtin_prefetch(xb + (r0 * 32 + c0), 0, 0);
    float lnll = 0.f;
    for (int py = 0; py < 4; ++py)
      for (int px = 0; px < 4; ++px) {
        float tv[16];
#pragma unroll
        for (int o = 0; o < 16; ++o) tv[o] = (float)t1b[b * 256 + o * 16 + py * 4 + px];
        for (int ky = 0; ky < 2; ++ky)
          for (int kx = 0; kx < 2; ++kx) {
            const int Y = py * 2 + ky, X = px * 2 + kx;
            for (int c = 0; c < 3; ++c) {
              float acc = bt2[c];
#pragma unroll
              for (int o = 0; o < 16; ++o)
                acc += tv[o] * sWt2[((o * 3 + c) * 2 + ky) * 2 + kx];
              const float v = sigmoidf_(acc);
              xrec[((((size_t)t * B_N + b) * 3 + c) * 8 + Y) * 8 + X] = v;
              const float d = v - xb[(c * 32 + r0 + Y) * 32 + c0 + X];
              lnll += d * d;
            }
          }
      }
    atomicAdd(nll, lnll);
  }
}

extern "C" void kernel_launch(void* const* d_in, const int* in_sizes, int n_in,
                              void* d_out, int out_size, void* d_ws, size_t ws_size,
                              hipStream_t stream) {
  const float* x    = (const float*)d_in[0];
  const float* act  = (const float*)d_in[1];
  const int*   pos  = (const int*)d_in[2];
  const float* s0   = (const float*)d_in[3];
  const float* nobs = (const float*)d_in[4];
  const float* nprd = (const float*)d_in[5];
  const float* eps  = (const float*)d_in[6];
  const float* W_a  = (const float*)d_in[7];
  const float* Ws1  = (const float*)d_in[8];
  const float* bs1  = (const float*)d_in[9];
  const float* Ws2  = (const float*)d_in[10];
  const float* bs2  = (const float*)d_in[11];
  const float* Wc1  = (const float*)d_in[12];
  const float* bc1  = (const float*)d_in[13];
  const float* g1   = (const float*)d_in[14];
  const float* be1  = (const float*)d_in[15];
  const float* Wc2  = (const float*)d_in[16];
  const float* bc2  = (const float*)d_in[17];
  const float* g2   = (const float*)d_in[18];
  const float* be2  = (const float*)d_in[19];
  const float* Wzm  = (const float*)d_in[20];
  const float* bzm  = (const float*)d_in[21];
  const float* Wzs  = (const float*)d_in[22];
  const float* bzs  = (const float*)d_in[23];
  const float* Wd1  = (const float*)d_in[24];
  const float* bd1  = (const float*)d_in[25];
  const float* gd1  = (const float*)d_in[26];
  const float* bed1 = (const float*)d_in[27];
  const float* Wd2  = (const float*)d_in[28];
  const float* bd2  = (const float*)d_in[29];
  const float* gd2  = (const float*)d_in[30];
  const float* bed2 = (const float*)d_in[31];
  const float* Wt1  = (const float*)d_in[32];
  const float* bt1  = (const float*)d_in[33];
  const float* gt1  = (const float*)d_in[34];
  const float* bet1 = (const float*)d_in[35];
  const float* Wt2  = (const float*)d_in[36];
  const float* bt2  = (const float*)d_in[37];

  float* out = (float*)d_out;
  float* st_obs  = out;                 // [256,256,2]
  float* st_pred = out + 131072;        // [32,256,2]
  float* zm_obs  = out + 147456;        // [256,256,16]
  float* zs_obs  = out + 1196032;       // [256,256,16]
  float* zm_pred = out + 2244608;       // [32,256,16]
  float* zs_pred = out + 2375680;       // [32,256,16]
  float* xrec    = out + 2506752;       // [32,256,3,8,8]
  float* nll     = out + 4079616;       // scalar

  k_zero_nll<<<1, 32, 0, stream>>>(nll);
  k_recur<<<1, 256, 0, stream>>>(act, s0, nobs, nprd, W_a, Ws1, bs1, Ws2, bs2, st_obs, st_pred);
  k_encoder<<<288, 256, 57088, stream>>>(x, pos, Wc1, bc1, g1, be1, Wc2, bc2, g2, be2,
                                         Wzm, bzm, Wzs, bzs, zm_obs, zs_obs, zm_pred, zs_pred);
  k_decoder<<<32, 256, 218368, stream>>>(zm_pred, zs_pred, eps, Wd1, bd1, gd1, bed1,
                                         Wd2, bd2, gd2, bed2, Wt1, bt1, gt1, bet1, Wt2, bt2,
                                         x, pos, xrec, nll);
}
